// AttentiveTransformer_37417755083485
// MI455X (gfx1250) — matrix operand order repaired and checked by reading, not compiled, not run on hardware
//
#include <hip/hip_runtime.h>
#include <stdint.h>

typedef __bf16 v16bf __attribute__((ext_vector_type(16)));
typedef float  v8f   __attribute__((ext_vector_type(8)));
typedef unsigned int u32x4 __attribute__((ext_vector_type(4)));
typedef unsigned int u32x8 __attribute__((ext_vector_type(8)));

#define D_DIM  512
#define F_DIM  512
#define MB     32                     // rows per block
#define KSTEP  32                     // K per WMMA step (bf16)
#define NKS    (D_DIM / KSTEP)        // 16 K-steps
#define NTILE  (F_DIM / 16)           // 32 N-tiles
#define WTILE_ELEMS (NTILE * 32 * 16) // 16384 bf16 (32 KB) per K-step, fragment-ordered
#define WTILE_DWORDS 8192             // 32 KB as 4-byte elements (for the TDM D#)

// GEMM staging and the h tile are live in disjoint phases -> union.
// af: A fragments  [2 slabs][32 lanes][16 bf16]           = 2 KB
// wf: B fragments  [2 bufs][32 tiles][32 lanes][16 bf16]  = 64 KB (double buffered)
// h : 32x512 f32 result tile                              = 64 KB
union SMem {
    struct {
        __bf16 af[2 * 32 * 16];
        __bf16 wf[2 * WTILE_ELEMS];
    } g;
    float h[MB * F_DIM];
};

// ---------------------------------------------------------------------------
// Prep: convert W (512x512 f32) once into bf16, laid out in B-fragment order:
//   wprep[ks][t][L][i] = W[ks*32 + (L/16)*16 + i][t*16 + (L%16)]
// so a WMMA B fragment is one contiguous 32-byte chunk per lane.
// ---------------------------------------------------------------------------
__global__ __launch_bounds__(256)
void prep_w_bf16(const float* __restrict__ W, __bf16* __restrict__ wprep)
{
    const int id  = blockIdx.x * 256 + threadIdx.x;  // 0..65535, 4 elems each
    const int c   = id & 511;                        // column of W
    const int k0g = (id >> 9) << 2;                  // first of 4 consecutive k
    const int ks  = k0g >> 5;
    const int kin = k0g & 31;
    const int L   = ((kin >> 4) << 4) + (c & 15);    // lane half by K, col in tile
    const int i0  = kin & 15;                        // fragment element (K = klo+i)
    const int t   = c >> 4;
    __bf16* dst = wprep + (((size_t)(ks * NTILE + t) * 32 + L) * 16 + i0);
    #pragma unroll
    for (int j = 0; j < 4; ++j)
        dst[j] = (__bf16)W[(size_t)(k0g + j) * F_DIM + c];
}

// ---------------------------------------------------------------------------
// Tensor Data Mover: one instruction DMAs a 32 KB fragment-ordered W tile
// into LDS. D# per CDNA5 ISA §8.3/8.4: user descriptor (count=1), 4-byte
// elements, 2-D tile 8192x1, tensor_dim0 = stride = 8192, groups 2/3 unused.
// Issued by one wave; tracked with TENSORcnt.
// ---------------------------------------------------------------------------
__device__ __forceinline__
void tdm_load_wtile(const __bf16* __restrict__ src, unsigned lds_addr)
{
    const unsigned long long ga = (unsigned long long)(uintptr_t)src;
    u32x4 g0;
    g0[0] = 0x1u;                                   // count=1 (valid user D#)
    g0[1] = lds_addr;                               // lds_addr[31:0] (bytes)
    g0[2] = (unsigned)(ga & 0xffffffffu);           // global_addr[31:0]
    g0[3] = (unsigned)((ga >> 32) & 0x01ffffffu)    // global_addr[56:32]
          | (2u << 30);                             // type=2 ("image")
    u32x8 g1;
    g1[0] = 0x00020000u;     // workgroup_mask=0, data_size=2 (4 B), no pad/iter
    g1[1] = (unsigned)(WTILE_DWORDS & 0xffff) << 16;   // tensor_dim0 lo16
    g1[2] = 0x00010000u;     // tensor_dim0 hi16 = 0, tensor_dim1 = 1 (lo16)
    g1[3] = (unsigned)(WTILE_DWORDS & 0xffff) << 16;   // tile_dim0 = 8192
    g1[4] = 0x00000001u;     // tile_dim1 = 1, tile_dim2 = 0
    g1[5] = (unsigned)WTILE_DWORDS;                    // tensor_dim0_stride lo32
    g1[6] = (unsigned)(WTILE_DWORDS & 0xffff) << 16;   // stride hi16=0, dim1_stride lo16
    g1[7] = 0u;                                        // tensor_dim1_stride hi32
    const u32x4 gz = (u32x4){0u, 0u, 0u, 0u};       // groups 2/3: dims 2-4 unused
    asm volatile("tensor_load_to_lds %0, %1, %2, %3"
                 :: "s"(g0), "s"(g1), "s"(gz), "s"(gz)
                 : "memory");
}

__global__ __launch_bounds__(256)
void fused_gemm_bn_sparsemax(const float* __restrict__ x,
                             const __bf16* __restrict__ wprep,
                             const float* __restrict__ bias,
                             const float* __restrict__ gamma,
                             const float* __restrict__ beta,
                             const float* __restrict__ mmean,
                             const float* __restrict__ mvar,
                             float* __restrict__ out)
{
    __shared__ SMem smem;

    const int tid     = threadIdx.x;
    const int lane    = tid & 31;          // wave32
    const int wave    = tid >> 5;          // 0..7
    const int half    = lane >> 4;
    const int lcol    = lane & 15;
    const int wave_m  = wave & 1;          // 16-row slab
    const int wave_ng = wave >> 1;         // group of 8 N-tiles
    const int row0    = blockIdx.x * MB;

    // LDS byte addresses for the TDM D# (flat LDS addr low 32 bits == LDS offset)
    const unsigned wf_lds0 = (unsigned)(uintptr_t)(void*)&smem.g.wf[0];
    const unsigned wf_lds1 = (unsigned)(uintptr_t)(void*)&smem.g.wf[WTILE_ELEMS];

    v8f acc[8];
    #pragma unroll
    for (int t = 0; t < 8; ++t)
        acc[t] = (v8f){0.f, 0.f, 0.f, 0.f, 0.f, 0.f, 0.f, 0.f};

    // Prologue: wave 0 starts streaming K-step 0 into buffer 0.
    if (wave == 0)
        tdm_load_wtile(wprep, wf_lds0);

    for (int ks = 0; ks < NKS; ++ks) {
        const int b = ks & 1;

        // ---- stage x tile (32x32 f32 -> bf16) directly in A-fragment order:
        //   element (r, k): half = k/16, L = half*16 + r%16,
        //   i = k%16
        {
            const int r     = tid >> 3;               // 0..31
            const int c4    = (tid & 7) << 2;         // 0..28
            const int wm    = r >> 4;
            const int rr    = r & 15;
            const int halfk = c4 >> 4;
            const int Lp    = halfk * 16 + rr;
            const int i0    = c4 & 15;
            const float4 v  = *(const float4*)(x + (size_t)(row0 + r) * D_DIM + ks * KSTEP + c4);
            __bf16* dst = &smem.g.af[(wm * 32 + Lp) * 16 + i0];
            dst[0] = (__bf16)v.x; dst[1] = (__bf16)v.y;
            dst[2] = (__bf16)v.z; dst[3] = (__bf16)v.w;
        }

        // ---- wave 0: prefetch next W tile into other buffer, drain current
        if (wave == 0) {
            if (ks + 1 < NKS) {
                tdm_load_wtile(wprep + (size_t)(ks + 1) * WTILE_ELEMS,
                               b ? wf_lds0 : wf_lds1);
                __builtin_amdgcn_s_wait_tensorcnt(1);   // current tile landed
            } else {
                __builtin_amdgcn_s_wait_tensorcnt(0);
            }
        }
        __syncthreads();

        // ---- fragments are contiguous 32 B per lane
        const v16bf afrag = *(const v16bf*)&smem.g.af[(wave_m * 32 + lane) * 16];
        #pragma unroll
        for (int t = 0; t < 8; ++t) {
            const int nt = wave_ng * 8 + t;
            const v16bf bfrag =
                *(const v16bf*)&smem.g.wf[b * WTILE_ELEMS + (nt * 32 + lane) * 16];
            acc[t] = __builtin_amdgcn_wmma_f32_16x16x32_bf16(
                false, afrag, false, bfrag, (short)0, acc[t], false, false);
        }
        __syncthreads();
    }

    // ---- BN epilogue folded as h = a*acc + c into the LDS h tile.
    #pragma unroll
    for (int t = 0; t < 8; ++t) {
        const int f = (wave_ng * 8 + t) * 16 + lcol;
        const float a = gamma[f] * rsqrtf(mvar[f] + 1e-5f);
        const float c = a * (bias[f] - mmean[f]) + beta[f];
        #pragma unroll
        for (int v = 0; v < 8; ++v) {
            const int row = wave_m * 16 + v + half * 8;
            smem.h[row * F_DIM + f] = acc[t][v] * a + c;
        }
    }
    __syncthreads();

    // ---- Row-wise sparsemax via Michelot's exact simplex projection.
    #pragma unroll 1
    for (int rr = 0; rr < 4; ++rr) {
        const int row = wave * 4 + rr;
        float z[16];
        #pragma unroll
        for (int j = 0; j < 16; ++j)
            z[j] = smem.h[row * F_DIM + lane + 32 * j];

        unsigned mask = 0xFFFFu;
        float tau = 0.f;
        for (int it = 0; it < F_DIM; ++it) {
            float s = 0.f; int c = 0;
            #pragma unroll
            for (int j = 0; j < 16; ++j)
                if ((mask >> j) & 1u) { s += z[j]; ++c; }
            #pragma unroll
            for (int off = 16; off > 0; off >>= 1) {
                s += __shfl_xor(s, off, 32);
                c += __shfl_xor(c, off, 32);
            }
            tau = (s - 1.0f) / (float)c;
            unsigned nm = 0u;
            #pragma unroll
            for (int j = 0; j < 16; ++j)
                nm |= (z[j] > tau) ? (1u << j) : 0u;
            const bool changed = (nm != mask);
            mask = nm;
            if (__ballot(changed) == 0ull) break;
        }

        #pragma unroll
        for (int j = 0; j < 16; ++j)
            out[(size_t)(row0 + row) * F_DIM + lane + 32 * j] =
                fmaxf(0.f, z[j] - tau);
    }
}

extern "C" void kernel_launch(void* const* d_in, const int* in_sizes, int n_in,
                              void* d_out, int out_size, void* d_ws, size_t ws_size,
                              hipStream_t stream)
{
    const float* x     = (const float*)d_in[0];
    const float* W     = (const float*)d_in[1];
    const float* b     = (const float*)d_in[2];
    const float* gamma = (const float*)d_in[3];
    const float* beta  = (const float*)d_in[4];
    const float* mmean = (const float*)d_in[5];
    const float* mvar  = (const float*)d_in[6];
    float* out   = (float*)d_out;
    __bf16* wpre = (__bf16*)d_ws;                 // 512 KB fragment-ordered bf16 W

    hipLaunchKernelGGL(prep_w_bf16, dim3(256), dim3(256), 0, stream, W, wpre);

    const int Brows = in_sizes[0] / D_DIM;        // 65536
    hipLaunchKernelGGL(fused_gemm_bn_sparsemax, dim3(Brows / MB), dim3(256), 0, stream,
                       x, wpre, b, gamma, beta, mmean, mvar, out);
}
